// Decode_78572131713672
// MI455X (gfx1250) — compile-verified, matrix operand order checked by reading, not run on hardware
//
#include <hip/hip_runtime.h>
#include <math.h>

typedef __attribute__((ext_vector_type(16))) __bf16 v16bf;
typedef __attribute__((ext_vector_type(8)))  float  v8f;

#define Bb 128
#define Tt 512
#define Hh 256
#define LIN 200
#define SP 58

// fast device transcendentals: v_exp_f32 + v_rcp_f32, few live temps
__device__ __forceinline__ float fsigmoid(float x) {
  return __builtin_amdgcn_rcpf(1.f + __expf(-x));
}
__device__ __forceinline__ float ftanh(float x) {
  x = fminf(fmaxf(x, -15.f), 15.f);
  float e = __expf(-2.f * x);
  return (1.f - e) * __builtin_amdgcn_rcpf(1.f + e);
}

// ---------------------------------------------------------------- utilities
__global__ void cast_bf16_kernel(const float* __restrict__ in,
                                 __bf16* __restrict__ out, long n) {
  long i = (long)blockIdx.x * blockDim.x + threadIdx.x;
  long stride = (long)gridDim.x * blockDim.x;
  for (; i < n; i += stride) out[i] = (__bf16)in[i];
}

// Pack a weight matrix W[Nr x (>=colOff+Kr)] (row-major fp32) into WMMA
// B-fragment order: out[((kc*NT + nt)*32 + lane)*16 + e] = bf16(W[n][colOff+k])
// with k = 32*kc + (e&1) + 2*((e>>1)&3) + 16*((e>>3)&1) + 8*(lane>>4)
//      n = nt*16 + (lane&15)          (zero outside Kr/Nr)
__global__ void fragpack_kernel(const float* __restrict__ in,
                                __bf16* __restrict__ out,
                                int KC, int NT, int Kr, int Nr,
                                int ldin, int colOff) {
  long total = (long)KC * NT * 512;
  long i = (long)blockIdx.x * blockDim.x + threadIdx.x;
  long stride = (long)gridDim.x * blockDim.x;
  for (; i < total; i += stride) {
    int e = (int)(i & 15);
    int lane = (int)((i >> 4) & 31);
    long fi = i >> 9;
    int nt = (int)(fi % NT);
    int kc = (int)(fi / NT);
    int k = 32 * kc + (e & 1) + 2 * ((e >> 1) & 3) + 16 * ((e >> 3) & 1) +
            8 * (lane >> 4);
    int n = nt * 16 + (lane & 15);
    float v = (k < Kr && n < Nr) ? in[(long)n * ldin + colOff + k] : 0.f;
    out[i] = (__bf16)v;
  }
}

// P[32][200] = pos_emb @ Wp^T ; L[8][200] = wordlen_emb @ Ww^T ; bias = b_ih+b_hh
__global__ void tables_kernel(const float* __restrict__ pos_emb,
                              const float* __restrict__ wl_emb,
                              const float* __restrict__ fcW,
                              const float* __restrict__ b_ih,
                              const float* __restrict__ b_hh,
                              float* __restrict__ P, float* __restrict__ L,
                              float* __restrict__ bias) {
  int idx = blockIdx.x * blockDim.x + threadIdx.x;
  if (idx < 32 * LIN) {
    int p = idx / LIN, j = idx % LIN;
    float s = 0.f;
    for (int d = 0; d < 50; ++d) s += pos_emb[p * 50 + d] * fcW[j * 582 + d];
    P[idx] = s;
  } else if (idx < 32 * LIN + 8 * LIN) {
    int i2 = idx - 32 * LIN;
    int w = i2 / LIN, j = i2 % LIN;
    float s = 0.f;
    for (int d = 0; d < 20; ++d) s += wl_emb[w * 20 + d] * fcW[j * 582 + 562 + d];
    L[i2] = s;
  } else if (idx < 32 * LIN + 8 * LIN + 1024) {
    int j = idx - (32 * LIN + 8 * LIN);
    bias[j] = b_ih[j] + b_hh[j];
  }
}

// last_sep[b][t] = last index s < t with sep_mask[b][s] != 0 (0 at t=0)
__global__ void lastsep_kernel(const int* __restrict__ sep,
                               int* __restrict__ lastsep) {
  int b = threadIdx.x;
  if (b >= Bb) return;
  int last = 0;
  for (int t = 0; t < Tt; ++t) {
    lastsep[b * Tt + t] = last;
    if (sep[b * Tt + t] > 0) last = t;
  }
}

// in-place exclusive prefix-sum of Y over t (per batch, per dim), ld = 208
__global__ void prefix_kernel(float* __restrict__ U) {
  int b = blockIdx.x;
  int d = threadIdx.x;
  if (d >= LIN) return;
  float run = 0.f;
  long base = (long)b * Tt * 208;
  for (int t = 0; t < Tt; ++t) {
    long i = base + (long)t * 208 + d;
    float y = U[i];
    U[i] = run;
    run += y;
  }
}

// z[b,t,j<224] = tanh((U[t]-U[ls])/wlen + P[last_pos] + L[wlen_id] + fc_b)
__global__ void z_kernel(const float* __restrict__ U,
                         const int* __restrict__ lastsep,
                         const int* __restrict__ pos_ids,
                         const float* __restrict__ P,
                         const float* __restrict__ L,
                         const float* __restrict__ fcb,
                         __bf16* __restrict__ z) {
  int m = blockIdx.x;           // m = b*T + t
  int t = m & (Tt - 1);
  int b = m >> 9;
  int j = threadIdx.x;
  if (j >= 224) return;
  if (j >= LIN || t == 0) { z[(long)m * 224 + j] = (__bf16)0.f; return; }
  int ls = lastsep[m];
  int wlen = t - ls; if (wlen < 1) wlen = 1;
  int wid = wlen < 7 ? wlen : 7;
  int lp = pos_ids[b * Tt + ls];
  float u1 = U[(long)m * 208 + j];
  float u0 = U[((long)b * Tt + ls) * 208 + j];
  float val = (u1 - u0) / (float)wlen + P[lp * LIN + j] + L[wid * LIN + j] + fcb[j];
  z[(long)m * 224 + j] = (__bf16)ftanh(val);
}

// ------------------------------------------------- generic bf16 WMMA GEMM
// C[M x Ncols] = A[M x K] (row-major bf16, K split across A/A2) * W (frag-packed)
// block tile 64x64, 8 waves x two 16x16 WMMA tiles. A staged in LDS already in
// fragment order (one b128 store / two b128 reads); B frags loaded straight
// from the frag-packed global weights (32B/lane, coalesced).
template <int EPI>
__global__ __launch_bounds__(256) void gemm_bf16_kernel(
    const __bf16* __restrict__ A, int lda,
    const __bf16* __restrict__ A2, int lda2, int ksplit,
    const __bf16* __restrict__ BtF, int NT,
    float* __restrict__ C, int ldc, int Ncols, int K,
    const float* __restrict__ bias, const int* __restrict__ length) {
  __shared__ __align__(32) __bf16 AsF[4][32][16];
  const int tid = threadIdx.x;
  const int lane = tid & 31, wv = tid >> 5;
  const int l15 = lane & 15, lh = lane >> 4;
  const int mb = blockIdx.y * 64, nb = blockIdx.x * 64;
  const int ntile0 = nb >> 4;
  // A staging: row-major 8k chunk == 8 consecutive fragment elements
  const int arow = tid >> 2, akq = (tid & 3) * 8;
  const int s_mt = arow >> 4;
  const int s_lane = (arow & 15) + 16 * ((akq >> 3) & 1);
  const int s_eo = 8 * (akq >> 4);
  v8f acc0, acc1;
  for (int i = 0; i < 8; ++i) { acc0[i] = 0.f; acc1[i] = 0.f; }

  for (int kb = 0, kc = 0; kb < K; kb += 32, ++kc) {
    { // stage A tile into fragment-order LDS
      int gk = kb + akq;
      const __bf16* src = (gk < ksplit)
          ? A  + (long)(mb + arow) * lda  + gk
          : A2 + (long)(mb + arow) * lda2 + (gk - ksplit);
      *(uint4*)&AsF[s_mt][s_lane][s_eo] = *(const uint4*)src;
      if (kb + 32 < K) __builtin_prefetch((const void*)(src + 32), 0, 1);
    }
    __syncthreads();
    for (int half = 0; half < 2; ++half) {
      int tile = wv * 2 + half;
      int mt = tile >> 2, nt = tile & 3;
      v16bf a = *(const v16bf*)&AsF[mt][lane][0];
      const __bf16* bp = BtF + ((long)(kc * NT + (ntile0 + nt)) * 32 + lane) * 16;
      v16bf b = *(const v16bf*)bp;
      if (kb + 32 < K) __builtin_prefetch((const void*)(bp + (long)NT * 512), 0, 1);
      if (half == 0)
        acc0 = __builtin_amdgcn_wmma_f32_16x16x32_bf16(false, a, false, b,
                                                       (short)0, acc0, false, false);
      else
        acc1 = __builtin_amdgcn_wmma_f32_16x16x32_bf16(false, a, false, b,
                                                       (short)0, acc1, false, false);
    }
    __syncthreads();
  }
  // epilogue; D layout: lane -> N = lane&15, vgpr r -> M = r + 8*(lane>>4)
  for (int half = 0; half < 2; ++half) {
    v8f acc = half ? acc1 : acc0;
    int tile = wv * 2 + half;
    int mt = tile >> 2, nt = tile & 3;
    int col = nb + nt * 16 + l15;
    if (col >= Ncols) continue;
    for (int r = 0; r < 8; ++r) {
      int row = mb + mt * 16 + r + 8 * lh;
      float v = acc[r];
      if (EPI == 1) v += bias[col];
      if (EPI == 2) {
        int tcur = row & (Tt - 1), bidx = row >> 9;
        if (tcur == 0 && col == 0) v = -1e30f;   // forbid APP at t=0
        if (tcur >= length[bidx]) v = 0.f;       // pad steps emit zeros
      }
      C[(long)row * ldc + col] = v;
    }
  }
}

// ----------------------------------------------------------- LSTM recurrence
// 8 blocks x 1024 threads (32 waves); each block owns 16 batch rows for all
// 512 steps. Per-thread VGPR cap at this occupancy is 128: kc 0..2 (6 frags =
// 48 VGPRs) stay register-resident; kc 3..7 stream from the L2-resident
// frag-packed W_hh every step via global_load_b128 with immediate offsets —
// only the byte OFFSET is made opaque per-iteration, so address-space
// inference still yields global loads while LICM cannot hoist + spill them.
__global__ __launch_bounds__(1024) void lstm_kernel(
    const float* __restrict__ Zih, const __bf16* __restrict__ WhhF,
    __bf16* __restrict__ hs) {
  __shared__ float gates[16][1024];
  __shared__ __align__(32) __bf16 hbF[8][32][16];  // fragment-order h (K=256)
  const int tid = threadIdx.x;
  const int lane = tid & 31, wv = tid >> 5;        // 32 waves
  const int l15 = lane & 15, lh = lane >> 4;
  const int b0 = blockIdx.x * 16;
  const int j = tid & 255;        // hidden column owned by this thread
  const int rg = tid >> 8;        // row group (4 rows each)
  // fragment-order coordinates for writing h[r][j]
  const int kc_w = j >> 5;
  const int kk = j & 31;
  const int lh_w = (kk >> 3) & 1;
  const int e_w = (kk & 7) + 8 * (kk >> 4);

  // kc 0..2 register-resident (6 frags = 48 VGPRs)
  v16bf wf[3][2];
#pragma unroll
  for (int kc = 0; kc < 3; ++kc)
#pragma unroll
    for (int nt = 0; nt < 2; ++nt)
      wf[kc][nt] = *(const v16bf*)(
          WhhF + ((long)(kc * 64 + wv * 2 + nt) * 32 + lane) * 16);
  // base pointer for streamed kc 3..7 (constant offsets from here)
  const __bf16* wsbase = WhhF + ((long)(3 * 64 + wv * 2) * 32 + lane) * 16;

  for (int i = tid; i < 8 * 32 * 16; i += 1024)
    ((__bf16*)hbF)[i] = (__bf16)0.f;
  float c[4];
  for (int r = 0; r < 4; ++r) c[r] = 0.f;

  for (int t = 0; t < Tt; ++t) {
    __syncthreads();  // h writes visible; previous gate reads complete
    // opaque zero offset: keeps the kc 3..7 fragment loads inside the loop
    // (true L2 stream) while preserving global address-space inference
    long off0 = 0;
    asm volatile("" : "+v"(off0));
    const __bf16* wsp = wsbase + off0;
    v8f acc[2];
#pragma unroll
    for (int nt = 0; nt < 2; ++nt)
      for (int i = 0; i < 8; ++i) acc[nt][i] = 0.f;
#pragma unroll
    for (int kc = 0; kc < 3; ++kc) {
      v16bf a = *(const v16bf*)&hbF[kc][lane][0];
#pragma unroll
      for (int nt = 0; nt < 2; ++nt)
        acc[nt] = __builtin_amdgcn_wmma_f32_16x16x32_bf16(false, a, false,
                                                          wf[kc][nt], (short)0,
                                                          acc[nt], false, false);
    }
#pragma unroll
    for (int kc = 3; kc < 8; ++kc) {
      v16bf a = *(const v16bf*)&hbF[kc][lane][0];
#pragma unroll
      for (int nt = 0; nt < 2; ++nt) {
        // frag(kc,nt) at wsp + (kc-3)*64*512 + nt*512  (immediate offsets)
        v16bf b = *(const v16bf*)(wsp + (long)(kc - 3) * 32768 + nt * 512);
        acc[nt] = __builtin_amdgcn_wmma_f32_16x16x32_bf16(false, a, false, b,
                                                          (short)0, acc[nt],
                                                          false, false);
      }
    }
#pragma unroll
    for (int nt = 0; nt < 2; ++nt) {
      int col = (wv * 2 + nt) * 16 + l15;
#pragma unroll
      for (int r = 0; r < 8; ++r) gates[r + 8 * lh][col] = acc[nt][r];
    }
    __syncthreads();  // gates ready
#pragma unroll
    for (int i = 0; i < 4; ++i) {
      int r = rg * 4 + i;
      long base = ((long)(b0 + r) * Tt + t) * 1024;
      float gi = gates[r][j]       + Zih[base + j];
      float gf = gates[r][j + 256] + Zih[base + j + 256];
      float gg = gates[r][j + 512] + Zih[base + j + 512];
      float go = gates[r][j + 768] + Zih[base + j + 768];
      if (i == 0 && t + 1 < Tt)
        __builtin_prefetch((const void*)&Zih[base + 1024 + j], 0, 1);
      c[i] = fsigmoid(gf) * c[i] + fsigmoid(gi) * ftanh(gg);
      float h = fsigmoid(go) * ftanh(c[i]);
      hbF[kc_w][r + 16 * lh_w][e_w] = (__bf16)h;
      hs[((long)(b0 + r) * Tt + t) * Hh + j] = (__bf16)h;
    }
  }
}

// ---------------------------------------------------------------- launcher
extern "C" void kernel_launch(void* const* d_in, const int* in_sizes, int n_in,
                              void* d_out, int out_size, void* d_ws, size_t ws_size,
                              hipStream_t stream) {
  const float* enc     = (const float*)d_in[0];   // [B,T,512]
  const int*   sep     = (const int*)  d_in[1];   // [B,T]
  const int*   pos_ids = (const int*)  d_in[2];   // [B,T]
  const int*   length  = (const int*)  d_in[3];   // [B]
  const float* pos_emb = (const float*)d_in[4];   // [32,50]
  const float* wl_emb  = (const float*)d_in[5];   // [8,20]
  const float* fcW     = (const float*)d_in[6];   // [200,582]
  const float* fcb     = (const float*)d_in[7];   // [200]
  const float* W_ih    = (const float*)d_in[8];   // [1024,200]
  const float* W_hh    = (const float*)d_in[9];   // [1024,256]
  const float* b_ih    = (const float*)d_in[10];  // [1024]
  const float* b_hh    = (const float*)d_in[11];  // [1024]
  const float* cmbW    = (const float*)d_in[12];  // [58,768]
  float* out = (float*)d_out;                     // [B*T,58]

  const long M = (long)Bb * Tt;  // 65536
  char* p = (char*)d_ws;
  auto alloc = [&](size_t bytes) -> void* {
    void* r = (void*)p;
    p += (bytes + 255) & ~(size_t)255;
    return r;
  };
  __bf16* enc_bf = (__bf16*)alloc(M * 512 * 2);          // enc in bf16
  __bf16* WeF    = (__bf16*)alloc((size_t)16 * 16 * 512 * 2);  // fc_W[:,50:562] frags
  __bf16* WihF   = (__bf16*)alloc((size_t)7 * 64 * 512 * 2);   // W_ih frags (K pad 224)
  __bf16* WhhF   = (__bf16*)alloc((size_t)8 * 64 * 512 * 2);   // W_hh frags
  __bf16* CmbF   = (__bf16*)alloc((size_t)24 * 4 * 512 * 2);   // combine_W frags
  float*  Ptab   = (float*) alloc(32 * LIN * 4);
  float*  Ltab   = (float*) alloc(8 * LIN * 4);
  float*  bias   = (float*) alloc(1024 * 4);
  int*    lsep   = (int*)   alloc(M * 4);
  float*  U      = (float*) alloc(M * 208 * 4);          // Y then prefix sums
  __bf16* zbf    = (__bf16*)alloc(M * 224 * 2);
  float*  Zih    = (float*) alloc(M * 1024 * 4);
  __bf16* hsbf   = (__bf16*)alloc(M * 256 * 2);

  // prep
  cast_bf16_kernel<<<8192, 256, 0, stream>>>(enc, enc_bf, M * 512);
  fragpack_kernel<<<256, 256, 0, stream>>>(fcW,  WeF,  16, 16, 512, 200, 582, 50);
  fragpack_kernel<<<512, 256, 0, stream>>>(W_ih, WihF, 7, 64, 200, 1024, 200, 0);
  fragpack_kernel<<<512, 256, 0, stream>>>(W_hh, WhhF, 8, 64, 256, 1024, 256, 0);
  fragpack_kernel<<<96,  256, 0, stream>>>(cmbW, CmbF, 24, 4, 768, 58, 768, 0);
  tables_kernel<<<36, 256, 0, stream>>>(pos_emb, wl_emb, fcW, b_ih, b_hh, Ptab, Ltab, bias);
  lastsep_kernel<<<1, 128, 0, stream>>>(sep, lsep);

  // Y = enc @ We^T   (M x 200, K=512)
  gemm_bf16_kernel<0><<<dim3(4, M / 64), 256, 0, stream>>>(
      enc_bf, 512, enc_bf, 512, 512, WeF, 16, U, 208, 200, 512, nullptr, nullptr);
  // U = exclusive prefix over t
  prefix_kernel<<<Bb, 256, 0, stream>>>(U);
  // z = tanh(dU/wlen + tables + fc_b)
  z_kernel<<<(int)M, 256, 0, stream>>>(U, lsep, pos_ids, Ptab, Ltab, fcb, zbf);
  // Zih = z @ W_ih^T + (b_ih + b_hh)   (M x 1024, K=224)
  gemm_bf16_kernel<1><<<dim3(16, M / 64), 256, 0, stream>>>(
      zbf, 224, zbf, 224, 224, WihF, 64, Zih, 1024, 1024, 224, bias, nullptr);
  // sequential LSTM recurrence (register + true L2-streamed W_hh)
  lstm_kernel<<<8, 1024, 0, stream>>>(Zih, WhhF, hsbf);
  // logits = [hs | enc] @ combine_W^T with masks fused  (M x 58, K=768)
  gemm_bf16_kernel<2><<<dim3(1, M / 64), 256, 0, stream>>>(
      hsbf, 256, enc_bf, 512, 256, CmbF, 4, out, SP, SP, 768, nullptr, length);
}
